// DormandPrinceIntegrator_18648747999580
// MI455X (gfx1250) — matrix operand order repaired and checked by reading, not compiled, not run on hardware
//
#include <hip/hip_runtime.h>
#include <hip/hip_bf16.h>
#include <math.h>

typedef __attribute__((ext_vector_type(16))) __bf16 v16bf;
typedef __attribute__((ext_vector_type(8)))  float  v8f;
typedef __attribute__((ext_vector_type(4)))  float  f4;

#define PI_F      3.14159265358979323846f
#define TWO_PI_F  6.28318530717958647692f
#define INV_2PI_F 0.15915494309189533577f
#define TWO_LOG2E 2.88539008177792681472f
#define DT        0.01f

// ---------------- DP5 tableau, restructured coefficients ----------------
// Stage ST (0..5) builds (x_{ST+1}, v_{ST+1}); row 6 is the b-row (final update).
//   v_i = cv + dt*s_i*f - dt*(sum_l A[i][l] S_l)@W
//   x_i = wrap(cx + dt*s_i*cv + dt^2*u_i*f - dt^2*(sum_l rc[i][l] S_l)@W)
struct Coef {
  float A[7][6];
  float s[7];
  float u[7];
  float rc[7][6];
};

__host__ __device__ constexpr Coef makeCoef() {
  Coef c{};
  double A[7][6] = {
    {0, 0, 0, 0, 0, 0},
    {1.0/5, 0, 0, 0, 0, 0},
    {3.0/40, 9.0/40, 0, 0, 0, 0},
    {44.0/45, -56.0/15, 32.0/9, 0, 0, 0},
    {19372.0/6561, -25360.0/2187, 64448.0/6561, -212.0/729, 0, 0},
    {9017.0/3168, -355.0/33, 46732.0/5247, 49.0/176, -5103.0/18656, 0},
    {35.0/384, 0, 500.0/1113, 125.0/192, -2187.0/6784, 11.0/84}
  };
  double s[7];
  for (int i = 0; i < 7; ++i) { double t = 0; for (int j = 0; j < 6; ++j) t += A[i][j]; s[i] = t; }
  for (int i = 0; i < 7; ++i) {
    double t = 0; for (int j = 0; j < 6; ++j) t += A[i][j] * s[j];
    c.u[i] = (float)t;
    c.s[i] = (float)s[i];
    for (int l = 0; l < 6; ++l) {
      double r = 0; for (int j = 0; j < 6; ++j) r += A[i][j] * A[j][l];
      c.rc[i][l] = (float)r;
      c.A[i][l]  = (float)A[i][l];
    }
  }
  return c;
}

__device__ __forceinline__ v8f zero8() {
  v8f z = {0.f, 0.f, 0.f, 0.f, 0.f, 0.f, 0.f, 0.f};
  return z;
}

__device__ __forceinline__ v8f wmma_bf16(v16bf a, v16bf b, v8f c) {
  return __builtin_amdgcn_wmma_f32_16x16x32_bf16(false, a, false, b, (short)0, c, false, false);
}

__device__ __forceinline__ void loadRun8(const float* p, float* dst) {
  f4 a = *(const f4*)p;
  f4 b = *(const f4*)(p + 4);
#pragma unroll
  for (int i = 0; i < 4; ++i) { dst[i] = a[i]; dst[4 + i] = b[i]; }
}

__device__ __forceinline__ void storeRun8(float* p, const float* v) {
  f4 a, b;
#pragma unroll
  for (int i = 0; i < 4; ++i) { a[i] = v[i]; b[i] = v[4 + i]; }
  *(f4*)p = a;
  *(f4*)(p + 4) = b;
}

__device__ __forceinline__ v16bf toBf16(const float* v) {
  v16bf r;
#pragma unroll
  for (int e = 0; e < 16; ++e) r[e] = (__bf16)v[e];
  return r;
}

// wrap(p) = mod(p+pi, 2pi) - pi, via floor (3 VALU ops, no loops)
__device__ __forceinline__ float wrapT(float p) {
  float t = (p + PI_F) * INV_2PI_F;
  return p - TWO_PI_F * floorf(t);
}

// Branch-free tanh. gfx1250 has native V_TANH_F32; else exp2/rcp identity:
// tanh(x) = sign(x) * (1 - 2/(exp2(2*log2e*|x|) + 1))
__device__ __forceinline__ float fast_tanh(float x) {
#if __has_builtin(__builtin_amdgcn_tanhf)
  return __builtin_amdgcn_tanhf(x);
#else
  float ax = fabsf(x);
  float e  = __builtin_amdgcn_exp2f(ax * TWO_LOG2E);
  float r  = 1.0f - 2.0f * __builtin_amdgcn_rcpf(e + 1.0f);
  return copysignf(r, x);
#endif
}

// LDS layout (floats):
//   [0      .. 4095 ]  W B-fragments (bf16 [8 chunks][2 ntiles][32 lanes][16])
//   [4096   .. 8191 ]  U B-fragments (bf16, same shape)
//   [8192   .. 57343]  per-wave tiles: wave w at 8192 + w*12288: xt(4096) vt(4096) ft(4096)
//   [57344  .. 61439]  per-wave scratch: wave w at 57344 + w*1024: scrV(512) scrX(512)
//   [61440  .. 73727]  per-wave S buffers: wave w at 61440 + w*3072: 6 stages x 512 (16x32 f32)
#define LDS_FLOATS 73728

// One DP5 stage, fully specialized at compile time (ST = 0..5 stages, 6 = final b-row).
template <int ST>
__device__ __forceinline__ void runStage(float* xt, float* vt, float* ft,
                                         float* scrV, float* scrX, float* sbuf,
                                         const v16bf* wfrag, const v16bf* ufrag,
                                         int lane, int m, int hi) {
  constexpr Coef CF = makeCoef();
  constexpr bool isFinal = (ST == 6);
  constexpr bool hasR = (ST >= 2);  // rc row 1 is identically zero (A row 0 is zero)
  const float sdt  = DT * CF.s[ST];
  const float udt2 = DT * DT * CF.u[ST];

  // ---- build A-fragments  Ta = -dt*sum_l A[ST][l] S_l ,  Ra = -dt^2*sum_l rc[ST][l] S_l
  // S_l live in LDS row-major (16x32): load per-lane A-layout runs directly.
  v16bf Ta, Ra;
  if constexpr (ST > 0) {
    float ta[16] __attribute__((aligned(16)));
    float ra[16] __attribute__((aligned(16)));
    {  // l = 0 term: assignment (avoid un-foldable +0 FMAs)
      const float ca = -DT * CF.A[ST][0];
      const float cr = -DT * DT * CF.rc[ST][0];
      float sv[16] __attribute__((aligned(16)));
      const float* sp = sbuf + m * 32 + hi * 8;
      loadRun8(sp, sv);
      loadRun8(sp + 16, sv + 8);
#pragma unroll
      for (int e = 0; e < 16; ++e) {
        ta[e] = ca * sv[e];
        if constexpr (hasR) ra[e] = cr * sv[e];
      }
    }
#pragma unroll
    for (int l = 1; l < 6; ++l) {
      if (l < ST) {  // compile-time fold (ST, l constant after unroll)
        const float ca = -DT * CF.A[ST][l];
        const float cr = -DT * DT * CF.rc[ST][l];
        const float* sp = sbuf + l * 512 + m * 32 + hi * 8;
        float sv[16] __attribute__((aligned(16)));
        loadRun8(sp, sv);
        loadRun8(sp + 16, sv + 8);
#pragma unroll
        for (int e = 0; e < 16; ++e) {
          ta[e] += ca * sv[e];
          if constexpr (hasR) ra[e] += cr * sv[e];
        }
      }
    }
    Ta = toBf16(ta);
    if constexpr (hasR) Ra = toBf16(ra);
  }

  v8f P0 = zero8(), P1 = zero8(), Q0 = zero8(), Q1 = zero8();

  for (int c = 0; c < 8; ++c) {
    const int cb = c * 32;
    float corrV[16] __attribute__((aligned(16)));
    float corrX[16] __attribute__((aligned(16)));
    if constexpr (ST > 0) {
      // corrV = Ta@W_chunk (+ corrX = Ra@W_chunk) in C layout -> A layout via scratch
#pragma unroll
      for (int nt = 0; nt < 2; ++nt) {
        v16bf bw = wfrag[(c * 2 + nt) * 32 + lane];
        v8f dv = wmma_bf16(Ta, bw, zero8());
#pragma unroll
        for (int r = 0; r < 8; ++r) scrV[(r + 8 * hi) * 32 + nt * 16 + m] = dv[r];
        if constexpr (hasR) {
          v8f dx = wmma_bf16(Ra, bw, zero8());
#pragma unroll
          for (int r = 0; r < 8; ++r) scrX[(r + 8 * hi) * 32 + nt * 16 + m] = dx[r];
        }
      }
      loadRun8(scrV + m * 32 + hi * 8,      corrV);
      loadRun8(scrV + m * 32 + 16 + hi * 8, corrV + 8);
      if constexpr (hasR) {
        loadRun8(scrX + m * 32 + hi * 8,      corrX);
        loadRun8(scrX + m * 32 + 16 + hi * 8, corrX + 8);
      }
    }

    float x0[16] __attribute__((aligned(16)));
    float v0[16] __attribute__((aligned(16)));
    float f0[16] __attribute__((aligned(16)));
    loadRun8(xt + m * 256 + cb + hi * 8,      x0);
    loadRun8(xt + m * 256 + cb + 16 + hi * 8, x0 + 8);
    loadRun8(vt + m * 256 + cb + hi * 8,      v0);
    loadRun8(vt + m * 256 + cb + 16 + hi * 8, v0 + 8);
    if constexpr (ST > 0) {
      loadRun8(ft + m * 256 + cb + hi * 8,      f0);
      loadRun8(ft + m * 256 + cb + 16 + hi * 8, f0 + 8);
    }

    float xx[16] __attribute__((aligned(16)));
    float vv[16] __attribute__((aligned(16)));
#pragma unroll
    for (int e = 0; e < 16; ++e) {
      if constexpr (ST == 0) {
        // stage 1 of DP5 evaluates dynamics at raw (cx, cv)
        vv[e] = v0[e];
        xx[e] = x0[e];
      } else {
        vv[e] = v0[e] + sdt * f0[e] + corrV[e];
        float xi = x0[e] + sdt * v0[e];
        if constexpr (hasR) xi += udt2 * f0[e] + corrX[e];
        xx[e] = wrapT(xi);
      }
    }

    if constexpr (isFinal) {
      storeRun8(xt + m * 256 + cb + hi * 8,      xx);
      storeRun8(xt + m * 256 + cb + 16 + hi * 8, xx + 8);
      storeRun8(vt + m * 256 + cb + hi * 8,      vv);
      storeRun8(vt + m * 256 + cb + 16 + hi * 8, vv + 8);
    } else {
      v16bf Ax = toBf16(xx);
      v16bf Av = toBf16(vv);
#pragma unroll
      for (int nt = 0; nt < 2; ++nt) {
        v16bf bu = ufrag[(c * 2 + nt) * 32 + lane];
        v8f& Pr = nt ? P1 : P0;
        v8f& Qr = nt ? Q1 : Q0;
        Pr = wmma_bf16(Ax, bu, Pr);
        Qr = wmma_bf16(Av, bu, Qr);
      }
    }
  }  // chunk loop

  if constexpr (!isFinal) {
    // S_ST = tanh(P) * Q^2, stored row-major (16x32) in LDS from C layout
#pragma unroll
    for (int nt = 0; nt < 2; ++nt) {
      const v8f& Pr = nt ? P1 : P0;
      const v8f& Qr = nt ? Q1 : Q0;
#pragma unroll
      for (int r = 0; r < 8; ++r) {
        float q = Qr[r];
        sbuf[ST * 512 + (r + 8 * hi) * 32 + nt * 16 + m] = fast_tanh(Pr[r]) * q * q;
      }
    }
  }
}

__global__ void dp5_geodesic_kernel(const float* __restrict__ gx,
                                    const float* __restrict__ gv,
                                    const float* __restrict__ gf,
                                    const float* __restrict__ gU,
                                    const float* __restrict__ gW,
                                    const int*   __restrict__ gsteps,
                                    float* __restrict__ outx,
                                    float* __restrict__ outv) {
  extern __shared__ char smem[];
  float*  smemf  = (float*)smem;
  __bf16* wfragS = (__bf16*)smem;            // 8192 bf16
  __bf16* ufragS = (__bf16*)(smem + 16384);  // 8192 bf16

  const int tid  = threadIdx.x;
  const int w    = tid >> 5;
  const int lane = tid & 31;
  const int m    = lane & 15;
  const int hi   = lane >> 4;
  const long long row0 = (long long)blockIdx.x * 64;

  // ---- build bf16 B-fragments of U and W, already in WMMA B layout ----
  // B layout (32x16, 16-bit): lane L -> column N = L&15; element j -> K = (L>>4)*8 + j (j<8), 16 + (L>>4)*8 + (j-8) (j>=8)
  for (int t = tid; t < 8192; t += 128) {
    int j  = t & 15;
    int ln = (t >> 4) & 31;
    int nt = (t >> 9) & 1;
    int c  = t >> 10;
    int lhi = ln >> 4, lm = ln & 15;
    int kk = (j & 7) + lhi * 8 + (j >> 3) * 16;
    ufragS[t] = (__bf16)gU[(c * 32 + kk) * 32 + nt * 16 + lm];
    wfragS[t] = (__bf16)gW[kk * 256 + c * 32 + nt * 16 + lm];
  }

  // ---- stage x / v / force tiles into LDS (f32, row-major per wave) ----
  for (int t = tid; t < 4096; t += 128) {
    int flat = t * 4;
    int r = flat >> 8, col = flat & 255;
    int ww = r >> 4, lr = r & 15;
    long long g = (row0 + r) * 256 + col;
    f4 vx = *(const f4*)(gx + g);
    f4 vv = *(const f4*)(gv + g);
    f4 vf = *(const f4*)(gf + g);
    float* base = smemf + 8192 + ww * 12288 + lr * 256 + col;
    *(f4*)(base)        = vx;
    *(f4*)(base + 4096) = vv;
    *(f4*)(base + 8192) = vf;
  }
  __syncthreads();

  float* xt   = smemf + 8192 + w * 12288;
  float* vt   = xt + 4096;
  float* ft   = xt + 8192;
  float* scrV = smemf + 57344 + w * 1024;
  float* scrX = scrV + 512;
  float* sbuf = smemf + 61440 + w * 3072;
  const v16bf* wfrag = (const v16bf*)wfragS;  // index (c*2+nt)*32 + lane
  const v16bf* ufrag = (const v16bf*)ufragS;

  const int nsteps = gsteps[0];

  for (int step = 0; step < nsteps; ++step) {
    runStage<0>(xt, vt, ft, scrV, scrX, sbuf, wfrag, ufrag, lane, m, hi);
    runStage<1>(xt, vt, ft, scrV, scrX, sbuf, wfrag, ufrag, lane, m, hi);
    runStage<2>(xt, vt, ft, scrV, scrX, sbuf, wfrag, ufrag, lane, m, hi);
    runStage<3>(xt, vt, ft, scrV, scrX, sbuf, wfrag, ufrag, lane, m, hi);
    runStage<4>(xt, vt, ft, scrV, scrX, sbuf, wfrag, ufrag, lane, m, hi);
    runStage<5>(xt, vt, ft, scrV, scrX, sbuf, wfrag, ufrag, lane, m, hi);
    runStage<6>(xt, vt, ft, scrV, scrX, sbuf, wfrag, ufrag, lane, m, hi);
  }

  __syncthreads();

  // ---- write results: out = [x (B*D) | v (B*D)] ----
  for (int t = tid; t < 4096; t += 128) {
    int flat = t * 4;
    int r = flat >> 8, col = flat & 255;
    int ww = r >> 4, lr = r & 15;
    const float* base = smemf + 8192 + ww * 12288 + lr * 256 + col;
    long long g = (row0 + r) * 256 + col;
    *(f4*)(outx + g) = *(const f4*)(base);
    *(f4*)(outv + g) = *(const f4*)(base + 4096);
  }
}

extern "C" void kernel_launch(void* const* d_in, const int* in_sizes, int n_in,
                              void* d_out, int out_size, void* d_ws, size_t ws_size,
                              hipStream_t stream) {
  const float* gx = (const float*)d_in[0];
  const float* gv = (const float*)d_in[1];
  const float* gf = (const float*)d_in[2];
  const float* gU = (const float*)d_in[3];
  const float* gW = (const float*)d_in[4];
  const int*   gs = (const int*)d_in[5];

  float* out = (float*)d_out;
  const long long B = (long long)in_sizes[0] / 256;  // D = 256
  float* outx = out;
  float* outv = out + B * 256;

  const int blocks = (int)(B / 64);                  // 64 rows per 128-thread block
  const size_t lds_bytes = (size_t)LDS_FLOATS * sizeof(float);  // 288 KB

  dp5_geodesic_kernel<<<blocks, 128, lds_bytes, stream>>>(gx, gv, gf, gU, gW, gs,
                                                          outx, outv);
}
